// LabelQueryAttentionHead_77386720739583
// MI455X (gfx1250) — matrix operand order repaired and checked by reading, not compiled, not run on hardware
//
#include <hip/hip_runtime.h>

typedef __attribute__((ext_vector_type(16))) __bf16          v16bf;
typedef __attribute__((ext_vector_type(16))) unsigned short  v16u;
typedef __attribute__((ext_vector_type(8)))  unsigned        v8u;
typedef __attribute__((ext_vector_type(8)))  float           v8f;
typedef __attribute__((ext_vector_type(2)))  __bf16          v2bf;

constexpr int NB = 16, NT = 512, NH = 1024, NL = 8192, ND = 256;

__device__ __forceinline__ unsigned short f2bf_sw(float f) {
  unsigned u = __builtin_bit_cast(unsigned, f);
  u += 0x7FFFu + ((u >> 16) & 1u);          // round-to-nearest-even
  return (unsigned short)(u >> 16);
}

// pack two f32 -> two bf16 in one dword: v_cvt_pk_bf16_f32 on gfx1250
__device__ __forceinline__ unsigned pk2bf(float lo, float hi) {
#if __has_builtin(__builtin_amdgcn_cvt_pk_bf16_f32)
  v2bf r = __builtin_amdgcn_cvt_pk_bf16_f32(lo, hi);
  return __builtin_bit_cast(unsigned, r);
#elif defined(__AMDGCN__)
  unsigned r;
  asm("v_cvt_pk_bf16_f32 %0, %1, %2" : "=v"(r) : "v"(lo), "v"(hi));
  return r;
#else
  return (unsigned)f2bf_sw(lo) | ((unsigned)f2bf_sw(hi) << 16);
#endif
}

__device__ __forceinline__ v8f wmma_bf16(v16u a, v16u b, v8f c) {
  return __builtin_amdgcn_wmma_f32_16x16x32_bf16(
      false, __builtin_bit_cast(v16bf, a),
      false, __builtin_bit_cast(v16bf, b),
      (short)0, c, false, false);
}

// build a 16-element bf16 fragment from two contiguous 16B runs
__device__ __forceinline__ v16u frag2(const unsigned short* p) {
  uint4 lo = *(const uint4*)(p);
  uint4 hi = *(const uint4*)(p + 16);
  v8u a;
  a[0] = lo.x; a[1] = lo.y; a[2] = lo.z; a[3] = lo.w;
  a[4] = hi.x; a[5] = hi.y; a[6] = hi.z; a[7] = hi.w;
  return __builtin_bit_cast(v16u, a);
}

// ---------------------------------------------------------------------------
// Kernel 0: elementwise fp32 -> packed bf16 (X, Wk, Wv, queries)
// ---------------------------------------------------------------------------
__global__ __launch_bounds__(256) void cvt_kernel(const float* __restrict__ src,
                                                  unsigned* __restrict__ dst,
                                                  int n4) {
  int i = blockIdx.x * 256 + threadIdx.x;
  if (i < n4) {
    float4 f = ((const float4*)src)[i];
    uint2 p;
    p.x = pk2bf(f.x, f.y);
    p.y = pk2bf(f.z, f.w);
    ((uint2*)dst)[i] = p;
  }
}

// ---------------------------------------------------------------------------
// Kernel 1: K = X*Wk^T  (stored [B*T][D] bf16),  V^T = (X*Wv^T)^T ([B][D][T])
// grid (B*T/16, 4), block 256 (8 waves). Barrier-free: A-fragments loaded
// straight from pre-converted Xbf as two b128 runs per lane.
// ---------------------------------------------------------------------------
__global__ __launch_bounds__(256) void kv_kernel(
    const unsigned short* __restrict__ Xbf,
    const unsigned short* __restrict__ Wkbf,
    const unsigned short* __restrict__ Wvbf,
    unsigned short* __restrict__ Kbf, unsigned short* __restrict__ Vt) {
  const int tid   = threadIdx.x;
  const int lane  = tid & 31;
  const int wave  = tid >> 5;
  const int mtile = blockIdx.x;                    // over B*T/16 = 512
  const int wtile = blockIdx.y * 8 + wave;         // 0..31
  const int mat   = wtile >> 4;                    // 0=K, 1=V
  const int ntile = wtile & 15;
  const unsigned short* Wb = mat ? Wvbf : Wkbf;
  const int m16   = lane & 15;
  const int d     = ntile * 16 + m16;
  const int kk0   = (lane < 16) ? 0 : 16;
  const int kbase = (lane < 16) ? 0 : 8;

  const unsigned short* xp = &Xbf[(mtile * 16 + m16) * NH + kbase];
  const unsigned short* wp = &Wb[d * NH + kbase];

  v8f acc = {};
  for (int h0 = 0; h0 < NH; h0 += 32) {
    v16u a = frag2(xp + h0);                          // A: X tile, bf16
    v16u b = frag2(wp + h0);                          // B: W[d][h] two 16B runs
    acc = wmma_bf16(a, b, acc);
  }

  const int rowbase = mtile * 16;
  if (mat == 0) {
    #pragma unroll
    for (int r = 0; r < 8; ++r) {
      int m = r + ((lane >= 16) ? 8 : 0);
      Kbf[(rowbase + m) * ND + d] = (unsigned short)pk2bf(acc[r], acc[r]);
    }
  } else {
    int bb = rowbase >> 9;               // / T
    int tb = rowbase & (NT - 1);
    int base = bb * ND * NT + d * NT + tb + ((lane >= 16) ? 8 : 0);
    #pragma unroll
    for (int r = 0; r < 8; r += 2)
      *(unsigned*)(&Vt[base + r]) = pk2bf(acc[r], acc[r + 1]);
  }
}

// ---------------------------------------------------------------------------
// Kernel 2: fused  logits = softmax(Q K^T * scale + mask) ; ctx = attn V ;
//           out = <ctx, Wo> + bias.  grid (L/16, B), block 256 (8 waves).
// ---------------------------------------------------------------------------
__global__ __launch_bounds__(256) void attn_kernel(
    const unsigned short* __restrict__ Qbf, const int* __restrict__ mask,
    const unsigned short* __restrict__ Kbf, const unsigned short* __restrict__ Vt,
    const float* __restrict__ Wo, const float* __restrict__ bias,
    float* __restrict__ out) {
  __shared__ float lds_f[16 * NT];            // 32 KB: logits, reused as ctx
  __shared__ unsigned short lds_pbf[16 * NT]; // 16 KB: probs in WMMA A-layout
  __shared__ float redA[256];

  const int tid   = threadIdx.x;
  const int lane  = tid & 31;
  const int wave  = tid >> 5;
  const int l0    = blockIdx.x * 16;
  const int bb    = blockIdx.y;
  const int m16   = lane & 15;
  const int kk0   = (lane < 16) ? 0 : 16;
  const int kbase = (lane < 16) ? 0 : 8;

  // Preload 8 q A-fragments: two 16B bf16 runs per lane, no conversion
  v16u aq[8];
  #pragma unroll
  for (int kt = 0; kt < 8; ++kt)
    aq[kt] = frag2(&Qbf[(l0 + m16) * ND + kt * 32 + kbase]);

  // Phase 1: logits = Q K^T * (1/sqrt(D)), masked.  4 t-tiles per wave.
  for (int ti = 0; ti < 4; ++ti) {
    int tt = wave + ti * 8;                 // t-tile 0..31
    int t  = tt * 16 + m16;
    if (ti < 3)  // prefetch next t-tile's K row (global_prefetch_b8)
      __builtin_prefetch(&Kbf[(bb * NT + t + 128) * ND], 0, 3);
    v8f acc = {};
    #pragma unroll
    for (int kt = 0; kt < 8; ++kt) {
      v16u b = frag2(&Kbf[(bb * NT + t) * ND + kt * 32 + kbase]);
      acc = wmma_bf16(aq[kt], b, acc);
    }
    float mval = (mask[bb * NT + t] == 0) ? -3.0e38f : 0.0f;
    #pragma unroll
    for (int r = 0; r < 8; ++r) {
      int m = r + ((lane >= 16) ? 8 : 0);
      lds_f[m * NT + t] = acc[r] * 0.0625f + mval;
    }
  }
  __syncthreads();

  // Phase 2: row softmax over T; normalized probs written as packed bf16
  // into WMMA A-fragment layout: lds_pbf[kt*512 + lane*16 + j].
  {
    int row = tid >> 4, seg = tid & 15;     // seg == k-chunk index kt
    float* rowp = &lds_f[row * NT + seg * 32];
    float lmax = -3.4e38f;
    for (int j = 0; j < 32; ++j) lmax = fmaxf(lmax, rowp[j]);
    redA[row * 16 + seg] = lmax;
    __syncthreads();
    float rmax = -3.4e38f;
    for (int j = 0; j < 16; ++j) rmax = fmaxf(rmax, redA[row * 16 + j]);
    float lsum = 0.0f;
    for (int j = 0; j < 32; ++j) {
      float e = __expf(rowp[j] - rmax);
      rowp[j] = e;
      lsum += e;
    }
    __syncthreads();                        // redA reads done before overwrite
    redA[row * 16 + seg] = lsum;
    __syncthreads();
    float rsum = 0.0f;
    for (int j = 0; j < 16; ++j) rsum += redA[row * 16 + j];
    float inv = 1.0f / rsum;
    unsigned* pbf32 = (unsigned*)lds_pbf;
    #pragma unroll
    for (int j = 0; j < 32; j += 2) {       // pairs stay in same lane-run
      int tl = row + (((j >> 3) & 1) << 4);
      int tj = (j & 7) + ((j >= 16) ? 8 : 0);
      pbf32[(seg * 512 + tl * 16 + tj) >> 1] =
          pk2bf(rowp[j] * inv, rowp[j + 1] * inv);
    }
  }
  __syncthreads();

  // Phase 3: ctx = attn @ V  (2 d-tiles per wave). ctx reuses lds_f storage.
  float* lds_ctx = lds_f;
  for (int di = 0; di < 2; ++di) {
    int dt = wave + di * 8;
    int d  = dt * 16 + m16;
    v8f acc = {};
    #pragma unroll
    for (int kt = 0; kt < 16; ++kt) {
      v16u a = *(const v16u*)(&lds_pbf[kt * 512 + lane * 16]);
      v16u b = frag2(&Vt[bb * ND * NT + d * NT + kt * 32 + kbase]);
      acc = wmma_bf16(a, b, acc);
    }
    #pragma unroll
    for (int r = 0; r < 8; ++r) {
      int m = r + ((lane >= 16) ? 8 : 0);
      lds_ctx[m * ND + d] = acc[r];
    }
  }
  __syncthreads();

  // Phase 4: out[b][l] = <ctx[l], Wo[l]> + bias[l]
  {
    int row = tid >> 4, seg = tid & 15;
    const float* wp = &Wo[(l0 + row) * ND + seg * 16];
    const float* cp = &lds_ctx[row * ND + seg * 16];
    float s = 0.0f;
    #pragma unroll
    for (int j = 0; j < 16; ++j) s += cp[j] * wp[j];
    redA[row * 16 + seg] = s;
    __syncthreads();
    if (tid < 16) {
      float tot = 0.0f;
      for (int j = 0; j < 16; ++j) tot += redA[tid * 16 + j];
      out[bb * NL + l0 + tid] = tot + bias[l0 + tid];
    }
  }
}

extern "C" void kernel_launch(void* const* d_in, const int* in_sizes, int n_in,
                              void* d_out, int out_size, void* d_ws, size_t ws_size,
                              hipStream_t stream) {
  const float* X    = (const float*)d_in[0];
  const int*   mask = (const int*)d_in[1];
  const float* Q    = (const float*)d_in[2];
  const float* Wk   = (const float*)d_in[3];
  const float* Wv   = (const float*)d_in[4];
  const float* Wo   = (const float*)d_in[5];
  const float* bias = (const float*)d_in[6];
  float* out = (float*)d_out;

  // workspace layout (bf16 buffers)
  unsigned short* Kbf  = (unsigned short*)d_ws;                  // 4 MB
  unsigned short* Vt   = Kbf  + (size_t)NB * NT * ND;            // 4 MB
  unsigned short* Wkbf = Vt   + (size_t)NB * NT * ND;            // 0.5 MB
  unsigned short* Wvbf = Wkbf + (size_t)ND * NH;                 // 0.5 MB
  unsigned short* Qbf  = Wvbf + (size_t)ND * NH;                 // 4 MB
  unsigned short* Xbf  = Qbf  + (size_t)NL * ND;                 // 16 MB

  const int nW4 = ND * NH / 4;          // 65536
  const int nQ4 = NL * ND / 4;          // 524288
  const int nX4 = NB * NT * NH / 4;     // 2097152
  cvt_kernel<<<nW4 / 256, 256, 0, stream>>>(Wk, (unsigned*)Wkbf, nW4);
  cvt_kernel<<<nW4 / 256, 256, 0, stream>>>(Wv, (unsigned*)Wvbf, nW4);
  cvt_kernel<<<nQ4 / 256, 256, 0, stream>>>(Q,  (unsigned*)Qbf,  nQ4);
  cvt_kernel<<<nX4 / 256, 256, 0, stream>>>(X,  (unsigned*)Xbf,  nX4);

  kv_kernel<<<dim3(NB * NT / 16, 4), 256, 0, stream>>>(Xbf, Wkbf, Wvbf, Kbf, Vt);
  attn_kernel<<<dim3(NL / 16, NB), 256, 0, stream>>>(Qbf, mask, Kbf, Vt, Wo, bias, out);
}
